// PersonalizedPageRankGCN_64192581206382
// MI455X (gfx1250) — compile-verified, matrix operand order checked by reading, not run on hardware
//
#include <hip/hip_runtime.h>
#include <hip/hip_bf16.h>

typedef __attribute__((ext_vector_type(16))) _Float16 v16h;
typedef __attribute__((ext_vector_type(8)))  float    v8f;

#define NODE_FEAT 66

// ---------------------------------------------------------------------------
// Degree accumulation: deg[dst[e]] += 1 for real edges (self loop added later)
// ---------------------------------------------------------------------------
__global__ void ppr_gcn_deg_kernel(const int* __restrict__ dst, float* __restrict__ deg, int E) {
    int e = blockIdx.x * blockDim.x + threadIdx.x;
    if (e < E) atomicAdd(&deg[dst[e]], 1.0f);
}

// deg -> dinv = rsqrt(deg + 1)   (self loop guarantees deg > 0)
__global__ void ppr_gcn_dinv_kernel(float* __restrict__ deg, int N) {
    int i = blockIdx.x * blockDim.x + threadIdx.x;
    if (i < N) deg[i] = rsqrtf(deg[i] + 1.0f);
}

// ---------------------------------------------------------------------------
// Zero-pad x [N x 66] -> xpad [N x 96] so the WMMA K-loop needs no guards.
// ---------------------------------------------------------------------------
__global__ void ppr_gcn_pad_x(const float* __restrict__ x, float* __restrict__ xp, int N) {
    int t = blockIdx.x * blockDim.x + threadIdx.x;
    if (t >= N * 96) return;
    int i = t / 96;
    int f = t - i * 96;
    xp[t] = (f < NODE_FEAT) ? x[(long long)i * NODE_FEAT + f] : 0.0f;
}

// ---------------------------------------------------------------------------
// Pack weights W[Korig x Nout] (fp32, row-major) into the CDNA5 B-fragment
// layout in f16, zero-padded to Kpad:
//   P[((nt*S + s)*32 + lane)*16 + h],  S = Kpad/32
//   lane -> col n = nt*16 + (lane&15); kbase = lane<16 ? 0 : 8
//   half h -> k = s*32 + kbase + (h<8 ? h : h+8)
// ---------------------------------------------------------------------------
__global__ void ppr_gcn_pack_w(const float* __restrict__ W, _Float16* __restrict__ P,
                               int Korig, int Kpad, int Nout) {
    int total = Kpad * Nout;          // == NT*S*32*16
    int t = blockIdx.x * blockDim.x + threadIdx.x;
    if (t >= total) return;
    int S = Kpad >> 5;
    int h    = t & 15;
    int lane = (t >> 4) & 31;
    int rest = t >> 9;
    int s  = rest % S;
    int nt = rest / S;
    int kbase = (lane < 16) ? 0 : 8;
    int k = s * 32 + kbase + (h < 8 ? h : h + 8);
    int n = nt * 16 + (lane & 15);
    float v = (k < Korig) ? W[(long long)k * Nout + n] : 0.0f;
    P[t] = (_Float16)v;
}

// ---------------------------------------------------------------------------
// WMMA GEMM: H[M x NOUT] = X[M x K] @ W (packed), fp32 activations, f16
// compute, f32 accumulate.
//  - Packed weights staged once per block in LDS (<= 32 KB of 320 KB/WGP);
//    all waves share the same B fragments -> ds_load_b128 feed, low latency.
//  - Each wave owns one 16-row M-tile and NTW=4 column tiles (32 acc VGPRs),
//    leaving registers free to keep A/B fragments in flight.
//  - K, NOUT compile-time: all loops unroll, addresses constant, no guards.
// ---------------------------------------------------------------------------
template <int K, int NOUT, int NTW>
__global__ void ppr_gcn_gemm_wmma(const float* __restrict__ X,
                                  const _Float16* __restrict__ P,
                                  float* __restrict__ H, int M) {
    constexpr int S   = K >> 5;        // K-slabs of 32
    constexpr int NT  = NOUT >> 4;     // 16-wide column tiles total
    constexpr int WPM = NT / NTW;      // waves per M-tile (column groups)

    __shared__ __align__(32) _Float16 sP[K * NOUT];

    // cooperative copy of packed weights to LDS (16B per thread per step)
    for (int idx = threadIdx.x * 8; idx < K * NOUT; idx += blockDim.x * 8) {
        *(float4*)(sP + idx) = *(const float4*)(P + idx);
    }
    __syncthreads();

    const int wavesPerBlock = blockDim.x >> 5;
    const int gwave = blockIdx.x * wavesPerBlock + (threadIdx.x >> 5);
    const int lane  = threadIdx.x & 31;
    const int mt    = gwave / WPM;
    const int nth   = gwave - mt * WPM;     // which column group
    if (mt >= (M >> 4)) return;             // wave-uniform: EXEC all-1s for WMMA

    const int   row   = mt * 16 + (lane & 15);
    const int   kbase = (lane < 16) ? 0 : 8;
    const float* rp   = X + (long long)row * K;

    v8f acc[NTW];
#pragma unroll
    for (int j = 0; j < NTW; ++j) acc[j] = (v8f){};

#pragma unroll
    for (int s = 0; s < S; ++s) {
        const int kb = s * 32 + kbase;
        float4 a0 = *(const float4*)(rp + kb);
        float4 a1 = *(const float4*)(rp + kb + 4);
        float4 a2 = *(const float4*)(rp + kb + 16);
        float4 a3 = *(const float4*)(rp + kb + 20);
        v16h a;
        a[0] = (_Float16)a0.x;  a[1] = (_Float16)a0.y;
        a[2] = (_Float16)a0.z;  a[3] = (_Float16)a0.w;
        a[4] = (_Float16)a1.x;  a[5] = (_Float16)a1.y;
        a[6] = (_Float16)a1.z;  a[7] = (_Float16)a1.w;
        a[8] = (_Float16)a2.x;  a[9] = (_Float16)a2.y;
        a[10] = (_Float16)a2.z; a[11] = (_Float16)a2.w;
        a[12] = (_Float16)a3.x; a[13] = (_Float16)a3.y;
        a[14] = (_Float16)a3.z; a[15] = (_Float16)a3.w;
#pragma unroll
        for (int j = 0; j < NTW; ++j) {
            const int nt = nth * NTW + j;
            v16h b = *(const v16h*)(sP + ((nt * S + s) * 32 + lane) * 16);
            acc[j] = __builtin_amdgcn_wmma_f32_16x16x32_f16(
                false, a, false, b, (short)0, acc[j], false, false);
        }
    }

    const int mrow = mt * 16 + ((lane < 16) ? 0 : 8);
    const int colb = lane & 15;
#pragma unroll
    for (int j = 0; j < NTW; ++j) {
        const int nt = nth * NTW + j;
#pragma unroll
        for (int r = 0; r < 8; ++r) {
            H[(long long)(mrow + r) * NOUT + nt * 16 + colb] = acc[j][r];
        }
    }
}

// ---------------------------------------------------------------------------
// Self-loop init of the aggregation buffer: agg[i,f] = H[i,f] * dinv[i]^2
// ---------------------------------------------------------------------------
__global__ void ppr_gcn_selfloop_init(const float* __restrict__ H,
                                      const float* __restrict__ dinv,
                                      float* __restrict__ agg, int N, int F) {
    int t = blockIdx.x * blockDim.x + threadIdx.x;
    if (t >= N * F) return;
    int i = t / F;
    float di = dinv[i];
    agg[t] = H[t] * di * di;
}

// ---------------------------------------------------------------------------
// Edge scatter, one wave per edge: agg[dst,f] += H[src,f]*dinv[src]*dinv[dst]
// Edge metadata loaded once per wave; feature lanes coalesced; f32 atomics
// resolve in L2 (feature matrices are L2-resident in MI455X's 192 MB L2).
// ---------------------------------------------------------------------------
__global__ void ppr_gcn_edge_scatter(const float* __restrict__ H,
                                     const int* __restrict__ src,
                                     const int* __restrict__ dst,
                                     const float* __restrict__ dinv,
                                     float* __restrict__ agg,
                                     int E, int F) {
    int wave = (blockIdx.x * blockDim.x + threadIdx.x) >> 5;
    int lane = threadIdx.x & 31;
    if (wave >= E) return;
    int s = src[wave];
    int d = dst[wave];
    float w = dinv[s] * dinv[d];
    const float* hs = H + (long long)s * F;
    float* ad = agg + (long long)d * F;
    for (int f = lane; f < F; f += 32) {
        atomicAdd(&ad[f], hs[f] * w);
    }
}

// ---------------------------------------------------------------------------
// Bias + ReLU, with optional strided/offset destination (for the concat).
// ---------------------------------------------------------------------------
__global__ void ppr_gcn_bias_relu(const float* __restrict__ in,
                                  const float* __restrict__ b,
                                  float* __restrict__ out,
                                  int N, int F, int outStride, int outOffset) {
    int t = blockIdx.x * blockDim.x + threadIdx.x;
    if (t >= N * F) return;
    int i = t / F;
    int f = t - i * F;
    float v = in[t] + b[f];
    out[(long long)i * outStride + outOffset + f] = fmaxf(v, 0.0f);
}

// ---------------------------------------------------------------------------
// PPR branch: relu(x[:, 56:66] @ Wp[10x32] + bp) -> concat cols 64..95
// K=10 is too small for WMMA to pay off; 320 VALU FMA per node.
// ---------------------------------------------------------------------------
__global__ void ppr_gcn_ppr_kernel(const float* __restrict__ x,
                                   const float* __restrict__ Wp,
                                   const float* __restrict__ bp,
                                   float* __restrict__ D, int N) {
    int t = blockIdx.x * blockDim.x + threadIdx.x;
    if (t >= N * 32) return;
    int i = t >> 5;
    int j = t & 31;
    const float* xr = x + (long long)i * NODE_FEAT + (NODE_FEAT - 10);
    float acc = bp[j];
#pragma unroll
    for (int k = 0; k < 10; ++k) acc += xr[k] * Wp[k * 32 + j];
    D[(long long)i * 96 + 64 + j] = fmaxf(acc, 0.0f);
}

// ---------------------------------------------------------------------------
// Head: sigmoid(fused[i,:64] . Wr + br)
// ---------------------------------------------------------------------------
__global__ void ppr_gcn_head_kernel(const float* __restrict__ fused,
                                    const float* __restrict__ Wr,
                                    const float* __restrict__ br,
                                    float* __restrict__ out, int N) {
    int i = blockIdx.x * blockDim.x + threadIdx.x;
    if (i >= N) return;
    const float* fr = fused + (long long)i * 64;
    float acc = br[0];
#pragma unroll
    for (int k = 0; k < 64; ++k) acc += fr[k] * Wr[k];
    out[i] = 1.0f / (1.0f + __expf(-acc));
}

// ---------------------------------------------------------------------------
// Launcher
// Inputs (setup_inputs order):
//  0:x [N,66] 1:edge_index [2,E] 2:W1[66,128] 3:b1 4:W2[128,128] 5:b2
//  6:W3[128,64] 7:b3 8:Wp[10,32] 9:bp 10:Wf[96,64] 11:bf 12:Wr[64,1] 13:br
// ---------------------------------------------------------------------------
extern "C" void kernel_launch(void* const* d_in, const int* in_sizes, int n_in,
                              void* d_out, int out_size, void* d_ws, size_t ws_size,
                              hipStream_t stream) {
    const float* x   = (const float*)d_in[0];
    const int*   ei  = (const int*)d_in[1];   // JAX x64 disabled -> int32
    const float* W1  = (const float*)d_in[2];
    const float* b1  = (const float*)d_in[3];
    const float* W2  = (const float*)d_in[4];
    const float* b2  = (const float*)d_in[5];
    const float* W3  = (const float*)d_in[6];
    const float* b3  = (const float*)d_in[7];
    const float* Wp  = (const float*)d_in[8];
    const float* bp  = (const float*)d_in[9];
    const float* Wf  = (const float*)d_in[10];
    const float* bf  = (const float*)d_in[11];
    const float* Wr  = (const float*)d_in[12];
    const float* br  = (const float*)d_in[13];
    float* out = (float*)d_out;

    const int N = in_sizes[0] / NODE_FEAT;
    const int E = in_sizes[1] / 2;
    const int* src = ei;
    const int* dst = ei + E;

    // ---- workspace layout ----
    // packed f16 weights first (32B aligned), then fp32 buffers
    _Float16* P1 = (_Float16*)d_ws;            //  96*128 halves
    _Float16* P2 = P1 + 96 * 128;              // 128*128
    _Float16* P3 = P2 + 128 * 128;             // 128*64
    _Float16* Pf = P3 + 128 * 64;              //  96*64
    float* fbase = (float*)(Pf + 96 * 64);     // total halves = 43008 -> 16B aligned
    float* dinv  = fbase;                      // N
    float* xpad  = dinv + N;                   // N*96
    float* A     = xpad + (long long)N * 96;   // N*128  (GEMM output H)
    float* B     = A + (long long)N * 128;     // N*128
    float* C     = B + (long long)N * 128;     // N*128
    float* D     = C + (long long)N * 128;     // N*96   (concat [h | ppr])

    const int TB = 256;
    auto blocksFor = [](long long total, int tb) { return (int)((total + tb - 1) / tb); };

    // ---- weight packing + input padding ----
    ppr_gcn_pack_w<<<blocksFor(96 * 128, TB), TB, 0, stream>>>(W1, P1, NODE_FEAT, 96, 128);
    ppr_gcn_pack_w<<<blocksFor(128 * 128, TB), TB, 0, stream>>>(W2, P2, 128, 128, 128);
    ppr_gcn_pack_w<<<blocksFor(128 * 64, TB), TB, 0, stream>>>(W3, P3, 128, 128, 64);
    ppr_gcn_pack_w<<<blocksFor(96 * 64, TB), TB, 0, stream>>>(Wf, Pf, 96, 96, 64);
    ppr_gcn_pad_x<<<blocksFor((long long)N * 96, TB), TB, 0, stream>>>(x, xpad, N);

    // ---- normalization ----
    hipMemsetAsync(dinv, 0, (size_t)N * sizeof(float), stream);
    ppr_gcn_deg_kernel<<<blocksFor(E, TB), TB, 0, stream>>>(dst, dinv, E);
    ppr_gcn_dinv_kernel<<<blocksFor(N, TB), TB, 0, stream>>>(dinv, N);

    // waves per GEMM = (N/16) * (NOUT/64); blockDim 256 = 8 waves
    auto gemmBlocks = [&](int wpm) {
        long long waves = (long long)(N >> 4) * wpm;
        return (int)((waves + 7) / 8);
    };
    auto aggregate = [&](const float* Hp, float* aggp, int F) {
        ppr_gcn_selfloop_init<<<blocksFor((long long)N * F, TB), TB, 0, stream>>>(Hp, dinv, aggp, N, F);
        ppr_gcn_edge_scatter<<<blocksFor((long long)E * 32, TB), TB, 0, stream>>>(
            Hp, src, dst, dinv, aggp, E, F);
    };

    // ---- layer 1: 66(pad 96) -> 128 ----
    ppr_gcn_gemm_wmma<96, 128, 4><<<gemmBlocks(2), TB, 0, stream>>>(xpad, P1, A, N);
    aggregate(A, B, 128);
    ppr_gcn_bias_relu<<<blocksFor((long long)N * 128, TB), TB, 0, stream>>>(B, b1, B, N, 128, 128, 0);

    // ---- layer 2: 128 -> 128 ----
    ppr_gcn_gemm_wmma<128, 128, 4><<<gemmBlocks(2), TB, 0, stream>>>(B, P2, A, N);
    aggregate(A, C, 128);
    ppr_gcn_bias_relu<<<blocksFor((long long)N * 128, TB), TB, 0, stream>>>(C, b2, C, N, 128, 128, 0);

    // ---- layer 3: 128 -> 64 ----
    ppr_gcn_gemm_wmma<128, 64, 4><<<gemmBlocks(1), TB, 0, stream>>>(C, P3, A, N);
    aggregate(A, B, 64);
    // write h into concat buffer D[:, 0:64] (row stride 96)
    ppr_gcn_bias_relu<<<blocksFor((long long)N * 64, TB), TB, 0, stream>>>(B, b3, D, N, 64, 96, 0);

    // ---- PPR branch into D[:, 64:96] ----
    ppr_gcn_ppr_kernel<<<blocksFor((long long)N * 32, TB), TB, 0, stream>>>(x, Wp, bp, D, N);

    // ---- fusion: [N,96] @ Wf[96,64] + bf, relu ----
    ppr_gcn_gemm_wmma<96, 64, 4><<<gemmBlocks(1), TB, 0, stream>>>(D, Pf, A, N);
    ppr_gcn_bias_relu<<<blocksFor((long long)N * 64, TB), TB, 0, stream>>>(A, bf, A, N, 64, 64, 0);

    // ---- head: sigmoid(A @ Wr + br) ----
    ppr_gcn_head_kernel<<<blocksFor(N, TB), TB, 0, stream>>>(A, Wr, br, out, N);
}
